// EdgeAwareMultiheadAttention_67997922230490
// MI455X (gfx1250) — matrix operand order repaired and checked by reading, not compiled, not run on hardware
//
#include <hip/hip_runtime.h>
#include <hip/hip_bf16.h>

// ---------------- problem constants ----------------
#define BATCH 2
#define LSEQ  512
#define EMBED 512
#define HEADS 8
#define HDIM  64
#define EHID  512

typedef __bf16 bf16;
typedef __attribute__((ext_vector_type(16))) __bf16 v16bf;
typedef __attribute__((ext_vector_type(8)))  __bf16 v8bf;
typedef __attribute__((ext_vector_type(8)))  float  v8f;

// ---------------- WMMA helpers (gfx1250, wave32) ----------------
// D = A(16x32 bf16) x B(32x16 bf16) + C(16x16 f32)
__device__ __forceinline__ v8f wmma_bf16(v16bf a, v16bf b, v8f c) {
  return __builtin_amdgcn_wmma_f32_16x16x32_bf16(
      /*neg_a=*/false, a, /*neg_b=*/false, b,
      /*c_mod=*/(short)0, c, /*reuse_a=*/false, /*reuse_b=*/false);
}

// A fragment: row-major A, tile base = &A[m0*lda + k0], lda in elements.
// ISA layout (16-bit A 16x32): lane L -> M = L&15;
//   lanes 0-15 : elems 0..7 -> K=0..7,  elems 8..15 -> K=16..23
//   lanes 16-31: elems 0..7 -> K=8..15, elems 8..15 -> K=24..31
__device__ __forceinline__ v16bf load_a_frag(const bf16* __restrict__ a, int lda) {
  int lane  = threadIdx.x & 31;
  int m     = lane & 15;
  int khalf = (lane >> 4) << 3;              // 0 or 8
  const bf16* row = a + (size_t)m * lda;
  v8bf lo = *(const v8bf*)(row + khalf);     // K = khalf .. khalf+7   (16B load)
  v8bf hi = *(const v8bf*)(row + khalf + 16);// K = khalf+16 .. +23    (16B load)
  v16bf f;
#pragma unroll
  for (int i = 0; i < 8; ++i) { f[i] = lo[i]; f[i + 8] = hi[i]; }
  return f;
}

// B fragment from N-major storage Bt[n, k] (contiguous in K), base = &Bt[n0*ldb + k0].
// ISA layout (bf16 B 32x16): lane L -> N = L&15; lanes 0-15: elems 0..7 -> K=0..7,
// elems 8..15 -> K=16..23; lanes 16-31: K=8..15, K=24..31 (two 16B loads).
__device__ __forceinline__ v16bf load_b_frag(const bf16* __restrict__ bt, int ldb) {
  int lane  = threadIdx.x & 31;
  int n     = lane & 15;
  int kbase = (lane >> 4) << 3;              // 0 or 8
  const bf16* col = bt + (size_t)n * ldb;
  v8bf lo = *(const v8bf*)(col + kbase);
  v8bf hi = *(const v8bf*)(col + kbase + 16);
  v16bf f;
#pragma unroll
  for (int i = 0; i < 8; ++i) { f[i] = lo[i]; f[i + 8] = hi[i]; }
  return f;
}

// C/D tile (16x16 f32, 8 VGPRs): lane L -> N = L&15, VGPR r -> M = r + 8*(L>=16)

// ---------------- elementwise conversions ----------------
__global__ void cvt_f32_bf16_kernel(const float* __restrict__ src,
                                    bf16* __restrict__ dst, int n) {
  int i = blockIdx.x * blockDim.x + threadIdx.x;
  if (i < n) dst[i] = (bf16)src[i];
}

// W [K,N] f32 -> Wt [N,K] bf16  (B-fragment friendly: contiguous along K)
__global__ void transpose_w_kernel(const float* __restrict__ w,
                                   bf16* __restrict__ wt, int K, int N) {
  int i = blockIdx.x * blockDim.x + threadIdx.x;
  if (i < K * N) {
    int k = i / N, n = i - k * N;
    wt[(size_t)n * K + k] = (bf16)w[i];
  }
}

// We2 [512, 8] f32 -> We2t [16, 512] bf16, N padded 8 -> 16 with zeros.
__global__ void prep_we2t_kernel(const float* __restrict__ We2,
                                 bf16* __restrict__ We2t) {
  int i = blockIdx.x * blockDim.x + threadIdx.x;   // 16*512
  if (i < 16 * EHID) {
    int n = i >> 9, k = i & 511;
    We2t[i] = (bf16)(n < HEADS ? We2[k * HEADS + n] : 0.f);
  }
}

// ---------------- projection GEMM: Out = (X @ W + bias) * scale ----------------
// One wave computes a 16x64 output strip: A fragment reused across 4 N-tiles.
// X: [1024, 512] bf16 row-major, Wt: [512(N), 512(K)] bf16.
// vtranspose==0 -> Out bf16 [1024,512]; ==1 -> Out = Vt bf16 [B,H,Dh,L].
__global__ void proj_gemm_kernel(const bf16* __restrict__ X,
                                 const bf16* __restrict__ Wt,
                                 const float* __restrict__ bias,
                                 float scale, bf16* __restrict__ Out,
                                 int vtranspose) {
  int t = blockIdx.x * (blockDim.x >> 5) + (threadIdx.x >> 5); // 512 waves
  int lane = threadIdx.x & 31;
  int ntq = t & 7, mt = t >> 3;            // 64 M-tiles x 8 N-quads
  int m0 = mt << 4, n0 = ntq << 6;
  const bf16* Ab = X + (size_t)m0 * EMBED;
  v8f c[4] = {};
  for (int k0 = 0; k0 < EMBED; k0 += 32) {
    __builtin_prefetch(Ab + k0 + 64, 0, 0);          // global_prefetch_b8
    v16bf a = load_a_frag(Ab + k0, EMBED);
#pragma unroll
    for (int q = 0; q < 4; ++q) {                    // A reuse across 4 N-tiles
      v16bf b = load_b_frag(Wt + (size_t)(n0 + (q << 4)) * EMBED + k0, EMBED);
      c[q] = wmma_bf16(a, b, c[q]);
    }
  }
  int mb = m0 + ((lane >> 4) << 3);
#pragma unroll
  for (int q = 0; q < 4; ++q) {
    int n = n0 + (q << 4) + (lane & 15);
    float bv = bias[n];
#pragma unroll
    for (int r = 0; r < 8; ++r) {
      float v = (c[q][r] + bv) * scale;
      int m = mb + r;
      if (!vtranspose) {
        Out[(size_t)m * EMBED + n] = (bf16)v;
      } else {                                 // Vt[b, h, d, l]
        int b_ = m >> 9, l = m & 511;
        int h  = n >> 6, d = n & 63;
        Out[(((size_t)(b_ * HEADS + h)) * HDIM + d) * LSEQ + l] = (bf16)v;
      }
    }
  }
}

// ---------------- edge MLP bias via WMMA ----------------
// One wave owns 16 edges (one 16-wide j-strip). Each lane builds the bf16
// A-fragment of H = relu(e*w1 + b1) directly in the ISA register layout
// (lane -> edge row M, K indices per 16-bit A map), then 16 WMMA k-steps
// against the constant We2^T B-fragment compute all 8 head biases at once.
__global__ void edge_bias_wmma_kernel(const float* __restrict__ edge, // [B,L,L]
                                      const float* __restrict__ We1,  // [512]
                                      const float* __restrict__ be1,  // [512]
                                      const bf16*  __restrict__ We2t, // [16,512] bf16
                                      const float* __restrict__ be2,  // [8]
                                      float* __restrict__ S) {        // [B,H,L,L]
  __shared__ float w1[EHID], b1[EHID];                               // 4 KB LDS
  for (int i = threadIdx.x; i < EHID; i += blockDim.x) { w1[i] = We1[i]; b1[i] = be1[i]; }
  __syncthreads();
  int t = blockIdx.x * (blockDim.x >> 5) + (threadIdx.x >> 5);       // 32768 waves
  int lane = threadIdx.x & 31;
  int jt = t & 31, row = t >> 5;            // row = b*512 + i2, in [0,1024)
  int b = row >> 9, i2 = row & 511;
  // Lane holds edge M = lane&15 (same e in both lane halves).
  float e = edge[((size_t)row << 9) + (jt << 4) + (lane & 15)];
  int khalf = (lane >> 4) << 3;             // 0 or 8
  v8f c = {};
  for (int k0 = 0; k0 < EHID; k0 += 32) {   // 16 WMMA k-steps
    v16bf a;
#pragma unroll
    for (int i = 0; i < 8; ++i) {           // K = k0+khalf .. +7
      int k = k0 + khalf + i;
      a[i] = (bf16)fmaxf(fmaf(e, w1[k], b1[k]), 0.f);
    }
#pragma unroll
    for (int i = 0; i < 8; ++i) {           // K = k0+khalf+16 .. +23
      int k = k0 + khalf + 16 + i;
      a[i + 8] = (bf16)fmaxf(fmaf(e, w1[k], b1[k]), 0.f);
    }
    v16bf bb = load_b_frag(We2t + k0, EHID);
    c = wmma_bf16(a, bb, c);                // EXEC all-ones here
  }
  int n = lane & 15, mb = (lane >> 4) << 3;
  if (n < HEADS) {                          // predicate only guards stores
    float bv = be2[n];
    float* Sp = S + (((size_t)(b * HEADS + n)) * LSEQ + i2) * LSEQ + (jt << 4) + mb;
#pragma unroll
    for (int r = 0; r < 8; ++r) Sp[r] = c[r] + bv;   // 8 consecutive floats
  }
}

// ---------------- scores: S += Qs @ K^T  (C preloaded with edge bias) ----------------
// One wave computes a 16x64 score strip: Q fragment reused across 4 key-tiles.
// Q pre-scaled by 1/sqrt(Dh) in its projection epilogue, so D = A*B + C is exact.
__global__ void scores_gemm_kernel(const bf16* __restrict__ Qb,  // [B*L, 512] bf16
                                   const bf16* __restrict__ Kb,  // [B*L, 512] bf16
                                   float* __restrict__ S) {      // [B,H,L,L] f32
  int t = blockIdx.x * (blockDim.x >> 5) + (threadIdx.x >> 5);   // 4096 waves
  int lane = threadIdx.x & 31;
  int ktq = t & 7, qt = (t >> 3) & 31, bh = t >> 8;              // bh in [0,16)
  int b = bh >> 3, h = bh & 7;
  const bf16* Ab = Qb + ((size_t)(b * LSEQ + (qt << 4))) * EMBED + h * HDIM;
  float* Srow = S + (((size_t)bh * LSEQ) + (qt << 4)) * LSEQ;
  int n = lane & 15, mb = (lane >> 4) << 3;
  v8f c[4];
#pragma unroll
  for (int q = 0; q < 4; ++q) {                                  // edge bias -> C
    float* Sp = Srow + (((ktq << 2) + q) << 4);
#pragma unroll
    for (int r = 0; r < 8; ++r) c[q][r] = Sp[(size_t)(mb + r) * LSEQ + n];
  }
#pragma unroll
  for (int k0 = 0; k0 < HDIM; k0 += 32) {
    v16bf a = load_a_frag(Ab + k0, EMBED);
#pragma unroll
    for (int q = 0; q < 4; ++q) {                                // Q reuse x4
      const bf16* Bb = Kb + ((size_t)(b * LSEQ + ((((ktq << 2) + q)) << 4))) * EMBED + h * HDIM;
      v16bf bf = load_b_frag(Bb + k0, EMBED);
      c[q] = wmma_bf16(a, bf, c[q]);
    }
  }
#pragma unroll
  for (int q = 0; q < 4; ++q) {
    float* Sp = Srow + (((ktq << 2) + q) << 4);
#pragma unroll
    for (int r = 0; r < 8; ++r) Sp[(size_t)(mb + r) * LSEQ + n] = c[q][r];
  }
}

// ---------------- row softmax (wave32 per row, shuffle reductions) ----------------
__global__ void softmax_kernel(const float* __restrict__ S,      // [B*H*L, 512]
                               bf16* __restrict__ A) {           // [B*H*L, 512] bf16
  int row  = blockIdx.x * (blockDim.x >> 5) + (threadIdx.x >> 5);// 8192 rows
  int lane = threadIdx.x & 31;
  const float* sp = S + (size_t)row * LSEQ;
  float v[16], mx = -3.0e38f;
#pragma unroll
  for (int i = 0; i < 16; ++i) { v[i] = sp[lane + (i << 5)]; mx = fmaxf(mx, v[i]); }
#pragma unroll
  for (int off = 16; off; off >>= 1) mx = fmaxf(mx, __shfl_xor(mx, off, 32));
  float sum = 0.f;
#pragma unroll
  for (int i = 0; i < 16; ++i) { v[i] = __expf(v[i] - mx); sum += v[i]; }
#pragma unroll
  for (int off = 16; off; off >>= 1) sum += __shfl_xor(sum, off, 32);
  float inv = 1.0f / sum;
  bf16* ap = A + (size_t)row * LSEQ;
#pragma unroll
  for (int i = 0; i < 16; ++i) ap[lane + (i << 5)] = (bf16)(v[i] * inv);
}

// ---------------- attn @ V : O[b,l,h*64+d] = sum_k attn * V ----------------
__global__ void av_gemm_kernel(const bf16* __restrict__ At,      // [B,H,L,L] bf16
                               const bf16* __restrict__ Vt,      // [B,H,Dh,L] bf16
                               bf16* __restrict__ O) {           // [B*L, 512] bf16
  int t = blockIdx.x * (blockDim.x >> 5) + (threadIdx.x >> 5);   // 512 waves
  int lane = threadIdx.x & 31;
  int qt = t & 31, bh = t >> 5;
  int b = bh >> 3, h = bh & 7;
  const bf16* Ab = At + ((size_t)bh * LSEQ + (qt << 4)) * LSEQ;
  const bf16* Vb = Vt + (size_t)bh * HDIM * LSEQ;
  v8f c[4] = {};
  for (int k0 = 0; k0 < LSEQ; k0 += 32) {
    v16bf a = load_a_frag(Ab + k0, LSEQ);
#pragma unroll
    for (int ntile = 0; ntile < 4; ++ntile) {     // reuse A across 4 N-tiles (Dh=64)
      v16bf bf = load_b_frag(Vb + (size_t)(ntile << 4) * LSEQ + k0, LSEQ);
      c[ntile] = wmma_bf16(a, bf, c[ntile]);
    }
  }
  int n = lane & 15, mb = (lane >> 4) << 3;
#pragma unroll
  for (int ntile = 0; ntile < 4; ++ntile) {
    int d = (ntile << 4) + n;
#pragma unroll
    for (int r = 0; r < 8; ++r) {
      int l = (qt << 4) + mb + r;
      O[((size_t)(b * LSEQ + l)) * EMBED + h * HDIM + d] = (bf16)c[ntile][r];
    }
  }
}

// ---------------- output projection: d_out f32 = O @ Wo + bo ----------------
// Same 16x64-per-wave structure as proj_gemm, f32 epilogue.
__global__ void out_proj_kernel(const bf16* __restrict__ O,
                                const bf16* __restrict__ Wt,
                                const float* __restrict__ bias,
                                float* __restrict__ Out) {
  int t = blockIdx.x * (blockDim.x >> 5) + (threadIdx.x >> 5);   // 512 waves
  int lane = threadIdx.x & 31;
  int ntq = t & 7, mt = t >> 3;
  int m0 = mt << 4, n0 = ntq << 6;
  const bf16* Ab = O + (size_t)m0 * EMBED;
  v8f c[4] = {};
  for (int k0 = 0; k0 < EMBED; k0 += 32) {
    __builtin_prefetch(Ab + k0 + 64, 0, 0);
    v16bf a = load_a_frag(Ab + k0, EMBED);
#pragma unroll
    for (int q = 0; q < 4; ++q) {
      v16bf b = load_b_frag(Wt + (size_t)(n0 + (q << 4)) * EMBED + k0, EMBED);
      c[q] = wmma_bf16(a, b, c[q]);
    }
  }
  int mb = m0 + ((lane >> 4) << 3);
#pragma unroll
  for (int q = 0; q < 4; ++q) {
    int n = n0 + (q << 4) + (lane & 15);
    float bv = bias[n];
#pragma unroll
    for (int r = 0; r < 8; ++r)
      Out[(size_t)(mb + r) * EMBED + n] = c[q][r] + bv;
  }
}

// ---------------- host launcher ----------------
extern "C" void kernel_launch(void* const* d_in, const int* in_sizes, int n_in,
                              void* d_out, int out_size, void* d_ws, size_t ws_size,
                              hipStream_t stream) {
  const float* query = (const float*)d_in[0];
  const float* key_in = (const float*)d_in[1];
  const float* value = (const float*)d_in[2];
  const float* edge  = (const float*)d_in[3];
  const float* Wq = (const float*)d_in[4];  const float* bq = (const float*)d_in[5];
  const float* Wk = (const float*)d_in[6];  const float* bk = (const float*)d_in[7];
  const float* Wv = (const float*)d_in[8];  const float* bv = (const float*)d_in[9];
  const float* Wo = (const float*)d_in[10]; const float* bo = (const float*)d_in[11];
  const float* We1 = (const float*)d_in[12]; const float* be1 = (const float*)d_in[13];
  const float* We2 = (const float*)d_in[14]; const float* be2 = (const float*)d_in[15];
  float* out = (float*)d_out;

  char* ws = (char*)d_ws;
  size_t off = 0;
  auto nxt = [&](size_t bytes) {
    size_t r = off; off += (bytes + 255) & ~(size_t)255; return r;
  };
  const size_t ACT  = (size_t)BATCH * LSEQ * EMBED;      // 524288 elems
  const size_t WSZ  = (size_t)EMBED * EMBED;             // 262144 elems
  const size_t SSZ  = (size_t)BATCH * HEADS * LSEQ * LSEQ;

  bf16* qbf   = (bf16*)(ws + nxt(ACT * 2));
  bf16* kbf   = (bf16*)(ws + nxt(ACT * 2));
  bf16* vbf   = (bf16*)(ws + nxt(ACT * 2));
  bf16* wqt   = (bf16*)(ws + nxt(WSZ * 2));
  bf16* wkt   = (bf16*)(ws + nxt(WSZ * 2));
  bf16* wvt   = (bf16*)(ws + nxt(WSZ * 2));
  bf16* wot   = (bf16*)(ws + nxt(WSZ * 2));
  bf16* we2t  = (bf16*)(ws + nxt((size_t)16 * EHID * 2));// 16 KB
  bf16* Qbuf  = (bf16*)(ws + nxt(ACT * 2));
  bf16* Kbuf  = (bf16*)(ws + nxt(ACT * 2));
  bf16* Vtb   = (bf16*)(ws + nxt(ACT * 2));
  float* S    = (float*)(ws + nxt(SSZ * 4));             // 16 MB, lives in L2
  bf16* Abuf  = (bf16*)(ws + nxt(SSZ * 2));              // 8 MB
  bf16* Obuf  = (bf16*)(ws + nxt(ACT * 2));

  // 1) fp32 -> bf16 activations
  cvt_f32_bf16_kernel<<<(int)(ACT / 256), 256, 0, stream>>>(query,  qbf, (int)ACT);
  cvt_f32_bf16_kernel<<<(int)(ACT / 256), 256, 0, stream>>>(key_in, kbf, (int)ACT);
  cvt_f32_bf16_kernel<<<(int)(ACT / 256), 256, 0, stream>>>(value,  vbf, (int)ACT);
  // 2) weights -> bf16, [N,K] B-fragment-friendly layouts
  transpose_w_kernel<<<(int)(WSZ / 256), 256, 0, stream>>>(Wq, wqt, EMBED, EMBED);
  transpose_w_kernel<<<(int)(WSZ / 256), 256, 0, stream>>>(Wk, wkt, EMBED, EMBED);
  transpose_w_kernel<<<(int)(WSZ / 256), 256, 0, stream>>>(Wv, wvt, EMBED, EMBED);
  transpose_w_kernel<<<(int)(WSZ / 256), 256, 0, stream>>>(Wo, wot, EMBED, EMBED);
  prep_we2t_kernel<<<32, 256, 0, stream>>>(We2, we2t);
  // 3) projections (WMMA, 16x64 per wave). Q folds the 1/sqrt(64) scale.
  proj_gemm_kernel<<<64, 256, 0, stream>>>(qbf, wqt, bq, 0.125f, Qbuf, 0);
  proj_gemm_kernel<<<64, 256, 0, stream>>>(kbf, wkt, bk, 1.0f,   Kbuf, 0);
  proj_gemm_kernel<<<64, 256, 0, stream>>>(vbf, wvt, bv, 1.0f,   Vtb,  1);
  // 4) edge MLP bias into S — hidden relu in VALU, 512x8 contraction on WMMA
  edge_bias_wmma_kernel<<<4096, 256, 0, stream>>>(edge, We1, be1, we2t, be2, S);
  // 5) scores: S = Qs @ K^T + S (edge bias preloaded as WMMA C operand; 16x64/wave)
  scores_gemm_kernel<<<512, 256, 0, stream>>>(Qbuf, Kbuf, S);
  // 6) softmax rows -> bf16 attention probabilities
  softmax_kernel<<<1024, 256, 0, stream>>>(S, Abuf);
  // 7) attn @ V (WMMA), heads re-merged into [B*L, 512]
  av_gemm_kernel<<<64, 256, 0, stream>>>(Abuf, Vtb, Obuf);
  // 8) output projection (WMMA) with f32 epilogue into d_out
  out_proj_kernel<<<64, 256, 0, stream>>>(Obuf, wot, bo, out);
}
